// ModHyenaBlock_42365557407849
// MI455X (gfx1250) — matrix operand order repaired and checked by reading, not compiled, not run on hardware
//
#include <hip/hip_runtime.h>

// ---------------- Types ----------------
typedef __attribute__((ext_vector_type(16))) _Float16 v16h;
typedef __attribute__((ext_vector_type(8)))  float    v8f;

#define LSEQ 8192
#define DMOD 768
#define DFFN 3072
#define NBATCH 2

__device__ __forceinline__ float2 cmul(float2 a, float2 b) {
    return make_float2(a.x * b.x - a.y * b.y, a.x * b.y + a.y * b.x);
}

// ---------------- f32 -> f16 convert ----------------
__global__ __launch_bounds__(256) void k_cvt(const float* __restrict__ in,
                                             _Float16* __restrict__ out, long long n) {
    long long i = (long long)blockIdx.x * 256 + threadIdx.x;
    if (i < n) out[i] = (_Float16)in[i];
}

// ---------------- window scale: h[t,d] *= exp(-t*exp(a)) ----------------
__global__ __launch_bounds__(256) void k_window(float* __restrict__ h,
                                                const float* __restrict__ a) {
    long long i = (long long)blockIdx.x * 256 + threadIdx.x;
    if (i >= (long long)LSEQ * DMOD) return;
    int t = (int)(i / DMOD);
    float decay = __expf(a[0]);
    h[i] *= __expf(-(float)t * decay);
}

// ---------------- LayerNorm over D=768, one wave per row, vectorized ----------------
__global__ __launch_bounds__(256) void k_ln(const float* __restrict__ x,
                                            const float* __restrict__ g,
                                            const float* __restrict__ be,
                                            float* __restrict__ o32,
                                            _Float16* __restrict__ o16,
                                            int nrows) {
    int row  = blockIdx.x * 8 + (threadIdx.x >> 5);
    int lane = threadIdx.x & 31;
    if (row >= nrows) return;
    const float4* xr = (const float4*)(x + (size_t)row * DMOD);
    float4 v4[6];
    float s = 0.f, ss = 0.f;
#pragma unroll
    for (int j = 0; j < 6; j++) {
        float4 v = xr[lane + j * 32];
        v4[j] = v;
        s  += v.x + v.y + v.z + v.w;
        ss += v.x * v.x + v.y * v.y + v.z * v.z + v.w * v.w;
    }
#pragma unroll
    for (int o = 16; o >= 1; o >>= 1) { s += __shfl_xor(s, o); ss += __shfl_xor(ss, o); }
    float m    = s * (1.f / DMOD);
    float var  = ss * (1.f / DMOD) - m * m;
    float rstd = rsqrtf(var + 1e-5f);
#pragma unroll
    for (int j = 0; j < 6; j++) {
        int c = lane * 4 + j * 128;
        float4 gv = *(const float4*)(g + c);
        float4 bv = *(const float4*)(be + c);
        float4 y;
        y.x = (v4[j].x - m) * rstd * gv.x + bv.x;
        y.y = (v4[j].y - m) * rstd * gv.y + bv.y;
        y.z = (v4[j].z - m) * rstd * gv.z + bv.z;
        y.w = (v4[j].w - m) * rstd * gv.w + bv.w;
        if (o32) *(float4*)(o32 + (size_t)row * DMOD + c) = y;
        if (o16) {
            union { _Float16 h[4]; uint2 u; } p;
            p.h[0] = (_Float16)y.x; p.h[1] = (_Float16)y.y;
            p.h[2] = (_Float16)y.z; p.h[3] = (_Float16)y.w;
            *(uint2*)(o16 + (size_t)row * DMOD + c) = p.u;
        }
    }
}

// ---------------- 32x32 tiled transpose, optional residual (out layout) ----------------
__global__ __launch_bounds__(256) void k_transpose(const float* __restrict__ in,
                                                   const float* __restrict__ res,
                                                   float* __restrict__ out,
                                                   int R, int C) {
    __shared__ float tile[32][33];
    int c0 = blockIdx.x * 32, r0 = blockIdx.y * 32;
    int tx = threadIdx.x, ty = threadIdx.y;   // (32,8)
#pragma unroll
    for (int j = 0; j < 32; j += 8)
        tile[ty + j][tx] = in[(size_t)(r0 + ty + j) * C + c0 + tx];
    __syncthreads();
#pragma unroll
    for (int j = 0; j < 32; j += 8) {
        int c = c0 + ty + j, r = r0 + tx;
        float v = tile[tx][ty + j];
        if (res) v += res[(size_t)c * R + r];
        out[(size_t)c * R + r] = v;
    }
}

// ---------------- in-place radix-2 FFT, 8192 points, 512 threads ----------------
#define FFT_N 8192
#define FFT_LOG 13
#define FFT_T 512

__device__ void fft8192(float2* S, int tid, float dir) {
    for (int i = tid; i < FFT_N; i += FFT_T) {
        int r = (int)(__brev((unsigned)i) >> (32 - FFT_LOG));
        if (r > i) { float2 t = S[i]; S[i] = S[r]; S[r] = t; }
    }
    __syncthreads();
    for (int half = 1; half < FFT_N; half <<= 1) {
        float base = dir * 3.14159265358979323846f / (float)half;
        for (int i = tid; i < FFT_N / 2; i += FFT_T) {
            int j  = i & (half - 1);
            int p0 = ((i & ~(half - 1)) << 1) | j;
            int p1 = p0 + half;
            float sn, cs; __sincosf(base * (float)j, &sn, &cs);
            float2 w = make_float2(cs, sn);
            float2 u = S[p0];
            float2 v = cmul(w, S[p1]);
            S[p0] = make_float2(u.x + v.x, u.y + v.y);
            S[p1] = make_float2(u.x - v.x, u.y - v.y);
        }
        __syncthreads();
    }
}

// X[k] of 16384-pt rFFT from 8192-pt complex FFT of packed even/odd samples.
__device__ __forceinline__ float2 rfft_recomb(const float2* S, int k) {
    float2 Ck = S[k & (FFT_N - 1)];
    float2 Cm = S[(FFT_N - k) & (FFT_N - 1)];
    float2 A  = make_float2(Ck.x + Cm.x, Ck.y - Cm.y);  // Ck + conj(Cm)
    float2 Bv = make_float2(Ck.x - Cm.x, Ck.y + Cm.y);  // Ck - conj(Cm)
    float2 Fo = make_float2(0.5f * Bv.y, -0.5f * Bv.x); // -i/2 * Bv
    float sn, cs; __sincosf(-3.14159265358979f * (float)k * (1.f / FFT_N), &sn, &cs);
    float2 WF = cmul(make_float2(cs, sn), Fo);
    return make_float2(0.5f * A.x + WF.x, 0.5f * A.y + WF.y);
}

// filter spectra: per d, rfft_{16384}(padded hw[:,d]) -> Wfh[d][0..8192]
__global__ __launch_bounds__(FFT_T) void k_fft_filter(const float* __restrict__ ht,
                                                      float2* __restrict__ Wfh) {
    __shared__ float2 S[FFT_N];
    int d = blockIdx.x, tid = threadIdx.x;
    const float2* x = (const float2*)(ht + (size_t)d * LSEQ);
    for (int m = tid; m < FFT_N; m += FFT_T)
        S[m] = (m < LSEQ / 2) ? x[m] : make_float2(0.f, 0.f);
    __syncthreads();
    fft8192(S, tid, -1.f);
    float2* out = Wfh + (size_t)d * (FFT_N + 1);
    for (int k = tid; k <= FFT_N; k += FFT_T) out[k] = rfft_recomb(S, k);
}

// fused conv: per (b,d): rfft(LN(z) channel), * Wfh, irfft, keep first 8192
__global__ __launch_bounds__(FFT_T) void k_conv(const float* __restrict__ lnzt,
                                                const float2* __restrict__ Wfh,
                                                float* __restrict__ convt) {
    __shared__ float2 S[FFT_N];
    int d = blockIdx.x, b = blockIdx.y, tid = threadIdx.x;
    const float2* x = (const float2*)(lnzt + ((size_t)b * DMOD + d) * LSEQ);
    for (int m = tid; m < FFT_N; m += FFT_T)
        S[m] = (m < LSEQ / 2) ? x[m] : make_float2(0.f, 0.f);
    __syncthreads();
    fft8192(S, tid, -1.f);

    const float2* wf = Wfh + (size_t)d * (FFT_N + 1);
    float2 c2a[9], c2b[9];
    int cnt = 0;
    for (int k = tid; k <= FFT_N / 2; k += FFT_T) {
        float2 Xk = rfft_recomb(S, k);
        float2 Xm = rfft_recomb(S, FFT_N - k);
        float2 Pk = cmul(Xk, wf[k]);
        float2 Pm = cmul(Xm, wf[FFT_N - k]);
        float sn, cs; __sincosf(3.14159265358979f * (float)k * (1.f / FFT_N), &sn, &cs);
        // C2[k] = Fe + i*Fo ; Fe = 0.5(Pk+conj(Pm)); Fo = 0.5 e^{+i pi k/N}(Pk-conj(Pm))
        {
            float2 A  = make_float2(Pk.x + Pm.x, Pk.y - Pm.y);
            float2 Bv = make_float2(Pk.x - Pm.x, Pk.y + Pm.y);
            float2 Fo = cmul(make_float2(cs, sn), Bv);
            c2a[cnt] = make_float2(0.5f * A.x - 0.5f * Fo.y, 0.5f * A.y + 0.5f * Fo.x);
        }
        // C2[N-k], q=N-k: e^{+i pi q/N} = (-cs, sn)
        {
            float2 A  = make_float2(Pm.x + Pk.x, Pm.y - Pk.y);
            float2 Bv = make_float2(Pm.x - Pk.x, Pm.y + Pk.y);
            float2 Fo = cmul(make_float2(-cs, sn), Bv);
            c2b[cnt] = make_float2(0.5f * A.x - 0.5f * Fo.y, 0.5f * A.y + 0.5f * Fo.x);
        }
        cnt++;
    }
    __syncthreads();
    cnt = 0;
    for (int k = tid; k <= FFT_N / 2; k += FFT_T) {
        S[k] = c2a[cnt];
        if (k > 0 && k < FFT_N / 2) S[FFT_N - k] = c2b[cnt];
        cnt++;
    }
    __syncthreads();
    fft8192(S, tid, 1.f);
    float2* out = (float2*)(convt + ((size_t)b * DMOD + d) * LSEQ);
    const float inv = 1.0f / FFT_N;
    for (int m = tid; m < LSEQ / 2; m += FFT_T) {
        float2 c = S[m];
        out[m] = make_float2(c.x * inv, c.y * inv);  // even/odd real samples
    }
}

// ---------------- WMMA GEMM: C = A(f16 MxK) @ W(f16 KxN), block tile 128x128 ----------------
// 8 waves: 4 (M) x 2 (N); each wave owns 32x64 = 2x4 WMMA tiles of 16x16, K-step 32.
// Register-staged pipeline: global loads for step i+1 issued before the WMMAs of step i.
// MODE 0: out f16 = silu(acc + bias)   MODE 1: out f32 = acc + bias (+res)
template <int MODE>
__global__ __launch_bounds__(256) void k_gemm(const _Float16* __restrict__ A,
                                              const _Float16* __restrict__ W,
                                              const float* __restrict__ bias,
                                              const float* __restrict__ res,
                                              void* __restrict__ Cout,
                                              int M, int N, int K) {
    __shared__ _Float16 Ash[128 * 40];  // row-major [m][k], stride 40 halves (80B, 16B-aligned)
    __shared__ _Float16 Bsh[128 * 40];  // col-major [n][k], stride 40
    const int tid  = threadIdx.x;
    const int m0   = blockIdx.x * 128;
    const int n0   = blockIdx.y * 128;
    const int lane = tid & 31, wid = tid >> 5;
    const int wm = wid & 3, wn = wid >> 2;   // 4x2 wave grid
    const int lg = lane >> 4, lr = lane & 15;

    // per-thread staging coordinates (constant-stride pointer bumping, no in-loop muls)
    const int arow = tid >> 2, akv = tid & 3;              // A: rows arow & arow+64, 8-half chunk akv
    const int bk = tid >> 3, bn = (tid & 7) * 16;          // B: row k0+bk, 16 halves at n0+bn
    const uint4* pa0 = (const uint4*)(A + (size_t)(m0 + arow) * K + akv * 8);
    const uint4* pa1 = (const uint4*)(A + (size_t)(m0 + 64 + arow) * K + akv * 8);
    const uint4* pb  = (const uint4*)(W + (size_t)bk * N + n0 + bn);
    const size_t bstep = ((size_t)32 * N) >> 3;            // 32 K-rows in uint4 units

    union V16 { v16h v; float4 q[2]; };
    v8f acc[2][4] = {};

    // pre-load first tiles into registers
    uint4 ar0 = pa0[0], ar1 = pa1[0];
    uint4 br0 = pb[0],  br1 = pb[1];

    for (int k0 = 0; k0 < K; k0 += 32) {
        // store staged registers to LDS
        *(uint4*)(Ash + arow * 40 + akv * 8)        = ar0;
        *(uint4*)(Ash + (64 + arow) * 40 + akv * 8) = ar1;
        {
            union { uint4 u[2]; _Float16 h[16]; } bb;
            bb.u[0] = br0; bb.u[1] = br1;
#pragma unroll
            for (int j = 0; j < 16; j++) Bsh[(bn + j) * 40 + bk] = bb.h[j];
        }
        __syncthreads();

        // issue next tile's global loads (latency overlapped with WMMA below)
        if (k0 + 32 < K) {
            pa0 += 4; pa1 += 4; pb += bstep;
            ar0 = pa0[0]; ar1 = pa1[0];
            br0 = pb[0];  br1 = pb[1];
            __builtin_prefetch(pa0 + 4, 0, 3);   // 2 steps ahead, WGP-local -> global_prefetch_b8
            __builtin_prefetch(pb + bstep, 0, 3);
        }

        V16 af[2], bf[4];
#pragma unroll
        for (int mt = 0; mt < 2; mt++) {
            const _Float16* p = Ash + (wm * 32 + mt * 16 + lr) * 40 + lg * 8;
            af[mt].q[0] = *(const float4*)(p);
            af[mt].q[1] = *(const float4*)(p + 16);
        }
#pragma unroll
        for (int nt = 0; nt < 4; nt++) {
            const _Float16* p = Bsh + (wn * 64 + nt * 16 + lr) * 40 + lg * 8;
            bf[nt].q[0] = *(const float4*)(p);
            bf[nt].q[1] = *(const float4*)(p + 16);
        }
#pragma unroll
        for (int mt = 0; mt < 2; mt++)
#pragma unroll
            for (int nt = 0; nt < 4; nt++)
                acc[mt][nt] = __builtin_amdgcn_wmma_f32_16x16x32_f16(
                    false, af[mt].v, false, bf[nt].v, (short)0, acc[mt][nt], false, false);
        __syncthreads();
    }

#pragma unroll
    for (int mt = 0; mt < 2; mt++)
#pragma unroll
        for (int nt = 0; nt < 4; nt++) {
            int n = n0 + wn * 64 + nt * 16 + lr;
            float bb = bias[n];
#pragma unroll
            for (int r = 0; r < 8; r++) {
                int m = m0 + wm * 32 + mt * 16 + lg * 8 + r;
                float v = acc[mt][nt][r] + bb;
                if (MODE == 0) {
                    float sv = v / (1.f + __expf(-v));   // SiLU
                    ((_Float16*)Cout)[(size_t)m * N + n] = (_Float16)sv;
                } else {
                    if (res) v += res[(size_t)m * N + n];
                    ((float*)Cout)[(size_t)m * N + n] = v;
                }
            }
        }
}

// ---------------- launch ----------------
extern "C" void kernel_launch(void* const* d_in, const int* in_sizes, int n_in,
                              void* d_out, int out_size, void* d_ws, size_t ws_size,
                              hipStream_t stream) {
    const float* z     = (const float*)d_in[0];
    const float* pos   = (const float*)d_in[1];
    const float* a     = (const float*)d_in[2];
    const float* w1p   = (const float*)d_in[3];
    const float* b1p   = (const float*)d_in[4];
    const float* w2p   = (const float*)d_in[5];
    const float* b2p   = (const float*)d_in[6];
    const float* w1    = (const float*)d_in[7];
    const float* b1    = (const float*)d_in[8];
    const float* w2    = (const float*)d_in[9];
    const float* b2    = (const float*)d_in[10];
    const float* gamma = (const float*)d_in[11];
    const float* beta  = (const float*)d_in[12];
    float* out = (float*)d_out;

    const long long L = LSEQ, D = DMOD, DFF = DFFN, B = NBATCH;
    char* ws = (char*)d_ws;
    size_t off = 0;
    auto take = [&](size_t bytes) { size_t o = off; off += (bytes + 255) & ~(size_t)255; return o; };

    size_t o_posh = take((size_t)L * D * 2);          // pos f16
    size_t o_up   = take((size_t)L * DFF * 2);        // pos-FFN hidden f16
    size_t o_h    = take((size_t)L * D * 4);          // filter h f32
    size_t o_ht   = take((size_t)L * D * 4);          // h^T (D,L)
    size_t o_wfh  = take((size_t)D * (FFT_N + 1) * 8);// filter spectra
    size_t o_lnz  = take((size_t)B * L * D * 4);
    size_t o_lnzt = take((size_t)B * L * D * 4);
    size_t o_convt= take((size_t)B * L * D * 4);
    size_t o_z1   = take((size_t)B * L * D * 4);
    size_t o_ln2h = take((size_t)B * L * D * 2);
    size_t o_w1ph = take((size_t)D * DFF * 2);        // f16 weights
    size_t o_w2ph = take((size_t)DFF * D * 2);
    size_t o_w1h  = take((size_t)D * DFF * 2);
    size_t o_w2h  = take((size_t)DFF * D * 2);
    size_t o_u    = o_posh;  // reuse head region (dead by then); needs B*L*DFF*2 <= head size

    _Float16* posh = (_Float16*)(ws + o_posh);
    _Float16* up   = (_Float16*)(ws + o_up);
    float*    h    = (float*)(ws + o_h);
    float*    ht   = (float*)(ws + o_ht);
    float2*   Wfh  = (float2*)(ws + o_wfh);
    float*    lnz  = (float*)(ws + o_lnz);
    float*    lnzt = (float*)(ws + o_lnzt);
    float*    convt= (float*)(ws + o_convt);
    float*    z1   = (float*)(ws + o_z1);
    _Float16* ln2h = (_Float16*)(ws + o_ln2h);
    _Float16* w1ph = (_Float16*)(ws + o_w1ph);
    _Float16* w2ph = (_Float16*)(ws + o_w2ph);
    _Float16* w1h  = (_Float16*)(ws + o_w1h);
    _Float16* w2h  = (_Float16*)(ws + o_w2h);
    _Float16* u    = (_Float16*)(ws + o_u);

    const long long nw = (long long)D * DFF;  // 2.36M, divisible by 256

    // 0) weights + pos -> f16 (once per launch)
    k_cvt<<<(int)((L * D) / 256), 256, 0, stream>>>(pos, posh, L * D);
    k_cvt<<<(int)(nw / 256), 256, 0, stream>>>(w1p, w1ph, nw);
    k_cvt<<<(int)(nw / 256), 256, 0, stream>>>(w2p, w2ph, nw);
    k_cvt<<<(int)(nw / 256), 256, 0, stream>>>(w1, w1h, nw);
    k_cvt<<<(int)(nw / 256), 256, 0, stream>>>(w2, w2h, nw);
    // 2) up = silu(posh @ w1p + b1p)
    k_gemm<0><<<dim3((int)(L / 128), (int)(DFF / 128)), 256, 0, stream>>>(
        posh, w1ph, b1p, nullptr, up, (int)L, (int)DFF, (int)D);
    // 3) h = up @ w2p + b2p
    k_gemm<1><<<dim3((int)(L / 128), (int)(D / 128)), 256, 0, stream>>>(
        up, w2ph, b2p, nullptr, h, (int)L, (int)D, (int)DFF);
    // 4) h *= window(t)
    k_window<<<(int)((L * D) / 256), 256, 0, stream>>>(h, a);
    // 5) ht = h^T  (in (L,D) -> out (D,L))
    k_transpose<<<dim3((int)(D / 32), (int)(L / 32)), dim3(32, 8), 0, stream>>>(
        h, nullptr, ht, (int)L, (int)D);
    // 6) filter spectra
    k_fft_filter<<<(int)D, FFT_T, 0, stream>>>(ht, Wfh);
    // 7) lnz = LN(z)
    k_ln<<<(int)((B * L) / 8), 256, 0, stream>>>(z, gamma, beta, lnz, nullptr, (int)(B * L));
    // 8) per-batch transpose lnz -> lnzt
    for (int b = 0; b < NBATCH; b++)
        k_transpose<<<dim3((int)(D / 32), (int)(L / 32)), dim3(32, 8), 0, stream>>>(
            lnz + (size_t)b * L * D, nullptr, lnzt + (size_t)b * D * L, (int)L, (int)D);
    // 9) conv per (b,d)
    k_conv<<<dim3((int)D, NBATCH), FFT_T, 0, stream>>>(lnzt, Wfh, convt);
    // 10) z1 = convt^T + z   (in (D,L) -> out (L,D))
    for (int b = 0; b < NBATCH; b++)
        k_transpose<<<dim3((int)(L / 32), (int)(D / 32)), dim3(32, 8), 0, stream>>>(
            convt + (size_t)b * D * L, z + (size_t)b * L * D, z1 + (size_t)b * L * D,
            (int)D, (int)L);
    // 11) ln2h = LN(z1) as f16
    k_ln<<<(int)((B * L) / 8), 256, 0, stream>>>(z1, gamma, beta, nullptr, ln2h, (int)(B * L));
    // 12) u = silu(ln2h @ w1 + b1)
    k_gemm<0><<<dim3((int)(B * L / 128), (int)(DFF / 128)), 256, 0, stream>>>(
        ln2h, w1h, b1, nullptr, u, (int)(B * L), (int)DFF, (int)D);
    // 13) out = u @ w2 + b2 + z1
    k_gemm<1><<<dim3((int)(B * L / 128), (int)(D / 128)), 256, 0, stream>>>(
        u, w2h, b2, z1, out, (int)(B * L), (int)D, (int)DFF);

    (void)in_sizes; (void)n_in; (void)out_size; (void)ws_size;
}
